// PyramidWindowAttention_69973607187052
// MI455X (gfx1250) — matrix operand order repaired and checked by reading, not compiled, not run on hardware
//
#include <hip/hip_runtime.h>
#include <hip/hip_bf16.h>
#include <math.h>

typedef _Float16 half_t;
typedef __attribute__((ext_vector_type(16))) _Float16 v16h;
typedef __attribute__((ext_vector_type(8)))  _Float16 v8h;
typedef __attribute__((ext_vector_type(8)))  float    v8f;

#define DIMC 256
#define LL   2
#define HH   128
#define WW_  256
#define HWSZ (HH * WW_)          // 32768
#define NTOK (LL * HWSZ)         // 65536

// ---------------------------------------------------------------------------
// CDNA5 async global -> LDS copy (ASYNCcnt path), 16B per lane.
// Generic pointers to LDS carry the LDS byte offset in their low 32 bits.
// ---------------------------------------------------------------------------
__device__ inline void async_load_b128(void* lds_ptr, const void* gptr) {
  uint32_t lds = (uint32_t)(uintptr_t)lds_ptr;
  uint64_t ga  = (uint64_t)(uintptr_t)gptr;
  asm volatile("global_load_async_to_lds_b128 %0, %1, off"
               :: "v"(lds), "v"(ga) : "memory");
}
__device__ inline void wait_async0() {
  asm volatile("s_wait_asynccnt 0" ::: "memory");
}

// ---------------------------------------------------------------------------
// WMMA fragment loaders (ISA 7.12.2 layouts, 16-bit, 16x16x32)
// A 16x32: lane L -> row (L&15); halves [0..7]=K[g*8..], [8..15]=K[16+g*8..]
// B 32x16: lane L -> col (L&15); halves [0..7]=K[g*8..], [8..15]=K[16+g*8..]
// C 16x16: lane L -> col (L&15); vgpr v -> row v + 8*(L>>4)
// ---------------------------------------------------------------------------
__device__ inline v16h load_a_frag(const half_t* base, int rs) {
  int lane = threadIdx.x & 31;
  int row = lane & 15, g = lane >> 4;
  v8h lo = *(const v8h*)(base + row * rs + g * 8);
  v8h hi = *(const v8h*)(base + row * rs + 16 + g * 8);
  v16h r;
#pragma unroll
  for (int i = 0; i < 8; ++i) { r[i] = lo[i]; r[8 + i] = hi[i]; }
  return r;
}

__device__ inline v16h load_b_frag(const half_t* base, int rs) {
  int lane = threadIdx.x & 31;
  int col = lane & 15, g = lane >> 4;
  const half_t* p = base + col * rs + g * 8;
  v8h lo = *(const v8h*)p;
  v8h hi = *(const v8h*)(p + 16);
  v16h r;
#pragma unroll
  for (int i = 0; i < 8; ++i) { r[i] = lo[i]; r[8 + i] = hi[i]; }
  return r;
}

// ---------------------------------------------------------------------------
// x (L,C,H,W) f32  ->  xh (N, C) f16, LDS-tiled transpose
// ---------------------------------------------------------------------------
__global__ void pack_x(const float* __restrict__ x, half_t* __restrict__ xh) {
  int b = blockIdx.x;
  int ct = b & 3; b >>= 2;
  int hwt = b & 511; b >>= 9;
  int l = b;
  int hw0 = hwt * 64, c0 = ct * 64;
  __shared__ float tile[64][65];
  int t = threadIdx.x;
  int hwL = t & 63, cL = t >> 6;
#pragma unroll
  for (int p = 0; p < 16; ++p) {
    int c = c0 + cL + p * 4;
    tile[cL + p * 4][hwL] = x[((size_t)l * DIMC + c) * HWSZ + hw0 + hwL];
  }
  __syncthreads();
  int cL2 = t & 63, hwL2 = t >> 6;
#pragma unroll
  for (int p = 0; p < 16; ++p) {
    int hw = hw0 + hwL2 + p * 4;
    xh[((size_t)(l * HWSZ + hw)) * DIMC + c0 + cL2] = (half_t)tile[cL2][hwL2 + p * 4];
  }
}

// f32 (K,N) -> f16 transposed (N,K)
__global__ void cvt_f16_t(const float* __restrict__ s, half_t* __restrict__ d, int K, int N) {
  int i = blockIdx.x * 256 + threadIdx.x;
  if (i >= K * N) return;
  int n = i / K, k = i % K;
  d[i] = (half_t)s[(size_t)k * N + n];
}

template <int WH, int WWIN>
__global__ void build_bias(const float* __restrict__ pos, float* __restrict__ tab) {
  const int T = WH * WWIN;
  int idx = blockIdx.x * 256 + threadIdx.x;
  if (idx >= T * T) return;
  int i = idx / T, j = idx % T;
  int d0 = (i / WWIN) - (j / WWIN) + WH - 1;
  int d1 = (i % WWIN) - (j % WWIN) + WWIN - 1;
  tab[idx] = pos[d0 * (2 * WWIN - 1) + d1];
}

// ---------------------------------------------------------------------------
// f16 WMMA GEMM: C[MxN] = A[MxK] * Bt[NxK]^T (+bias)   BM=BN=128, BK=32
// 256 threads = 8 waves (4x2), wave -> 32x64; async-copy double-buffered LDS
// ---------------------------------------------------------------------------
__global__ void gemm_f16(const half_t* __restrict__ A, int lda,
                         const half_t* __restrict__ Bt, int ldb,
                         half_t* __restrict__ Cm, int ldc,
                         const float* __restrict__ bias, int K) {
  __shared__ __align__(16) half_t As[2][128 * 40];
  __shared__ __align__(16) half_t Bs[2][128 * 40];
  int tid = threadIdx.x;
  int m0 = blockIdx.y * 128, n0 = blockIdx.x * 128;
  int wid = tid >> 5, lane = tid & 31;
  int wm = wid >> 1, wn = wid & 1;

  v8f acc[2][4] = {};
  const int steps = K >> 5;

  auto stage = [&](int buf, int k0) {
#pragma unroll
    for (int i = tid; i < 512; i += 256) {
      int r = i >> 2, ck = (i & 3) * 8;
      async_load_b128(&As[buf][r * 40 + ck], A + (size_t)(m0 + r) * lda + k0 + ck);
      async_load_b128(&Bs[buf][r * 40 + ck], Bt + (size_t)(n0 + r) * ldb + k0 + ck);
    }
  };

  stage(0, 0);
  for (int s = 0; s < steps; ++s) {
    int cur = s & 1;
    wait_async0();
    __syncthreads();
    if (s + 1 < steps) stage(cur ^ 1, (s + 1) * 32);

    v16h af[2], bf[4];
#pragma unroll
    for (int i = 0; i < 2; ++i) af[i] = load_a_frag(&As[cur][(wm * 32 + i * 16) * 40], 40);
#pragma unroll
    for (int j = 0; j < 4; ++j) bf[j] = load_b_frag(&Bs[cur][(wn * 64 + j * 16) * 40], 40);
#pragma unroll
    for (int i = 0; i < 2; ++i)
#pragma unroll
      for (int j = 0; j < 4; ++j)
        acc[i][j] = __builtin_amdgcn_wmma_f32_16x16x32_f16(
            false, af[i], false, bf[j], (short)0, acc[i][j], false, false);
    __syncthreads();
  }

  int colL = lane & 15, hiL = lane >> 4;
#pragma unroll
  for (int i = 0; i < 2; ++i)
#pragma unroll
    for (int j = 0; j < 4; ++j) {
      int col = n0 + wn * 64 + j * 16 + colL;
      float bv = bias ? bias[col] : 0.f;
#pragma unroll
      for (int vg = 0; vg < 8; ++vg) {
        int m = m0 + wm * 32 + i * 16 + vg + 8 * hiL;
        Cm[(size_t)m * ldc + col] = (half_t)(acc[i][j][vg] + bv);
      }
    }
}

// ---------------------------------------------------------------------------
// Windowed attention, one workgroup per (l, window, head, row-block)
// async-staged Q/K/V; S = Q K^T * scale + bias ; softmax ; O = P V (WMMA)
// ---------------------------------------------------------------------------
template <int WH, int WWIN, int HEADS, int DH, int RB>
__global__ void win_attn(const half_t* __restrict__ qkv,
                         const float* __restrict__ biasTab,
                         half_t* __restrict__ attnout, float scale) {
  constexpr int T = WH * WWIN;
  constexpr int DHP = (DH < 32) ? 32 : DH;
  constexpr int TP = (T < 32) ? 32 : T;
  constexpr int NWW = WW_ / WWIN;
  constexpr int NW = (HH / WH) * NWW;
  constexpr int RBLKS = T / RB;
  constexpr int CH = DH / 8;   // 16B chunks per row

  extern __shared__ __align__(16) char smem[];
  float*  sS  = (float*)smem;                          // RB x T
  half_t* sQ  = (half_t*)(smem + (size_t)RB * T * 4);  // RB x DHP
  half_t* sK  = sQ + RB * DHP;                         // T x DHP
  half_t* sV  = sK + T * DHP;                          // T x DHP (row-major)
  half_t* sVt = sV + T * DHP;                          // DH x TP
  half_t* sP  = sVt + DH * TP;                         // RB x TP

  int b = blockIdx.x;
  int rb = b % RBLKS; b /= RBLKS;
  int head = b % HEADS; b /= HEADS;
  int win = b % NW; int l = b / NW;
  int bh = win / NWW, bw = win % NWW;
  int rb0 = rb * RB;
  int tid = threadIdx.x;

  // zero-fill (covers K-dim / token-dim padding)
  const int totH = RB * DHP + 2 * T * DHP + DH * TP + RB * TP;
  for (int i = tid; i < totH; i += 256) sQ[i] = (half_t)0.f;
  __syncthreads();

  // async stage K, V rows
  for (int i = tid; i < T * CH; i += 256) {
    int t = i / CH, ck = (i % CH) * 8;
    int h = bh * WH + t / WWIN, w = bw * WWIN + t % WWIN;
    size_t n = (size_t)(l * HH + h) * WW_ + w;
    const half_t* base = qkv + n * 768 + head * DH + ck;
    async_load_b128(&sK[t * DHP + ck], base + 256);
    async_load_b128(&sV[t * DHP + ck], base + 512);
  }
  // async stage Q rows of this row-block
  for (int i = tid; i < RB * CH; i += 256) {
    int r = i / CH, ck = (i % CH) * 8;
    int t = rb0 + r;
    int h = bh * WH + t / WWIN, w = bw * WWIN + t % WWIN;
    size_t n = (size_t)(l * HH + h) * WW_ + w;
    async_load_b128(&sQ[r * DHP + ck], qkv + n * 768 + head * DH + ck);
  }
  wait_async0();
  __syncthreads();

  // LDS->LDS transpose of V (B-frag for P*V needs V^T)
  for (int i = tid; i < T * DH; i += 256) {
    int t = i / DH, d = i % DH;
    sVt[d * TP + t] = sV[t * DHP + d];
  }
  __syncthreads();

  int wid = tid >> 5, lane = tid & 31;
  int colL = lane & 15, hiL = lane >> 4;

  // ---- S = Q K^T * scale + bias ----
  constexpr int RT = RB / 16, CTS = T / 16;
  for (int tile = wid; tile < RT * CTS; tile += 8) {
    int tr = tile % RT, tc = tile / RT;
    v8f acc = {};
#pragma unroll
    for (int kk = 0; kk < DHP; kk += 32) {
      v16h a = load_a_frag(sQ + tr * 16 * DHP + kk, DHP);
      v16h bb = load_b_frag(sK + tc * 16 * DHP + kk, DHP);
      acc = __builtin_amdgcn_wmma_f32_16x16x32_f16(false, a, false, bb, (short)0, acc, false, false);
    }
    int j = tc * 16 + colL;
#pragma unroll
    for (int vg = 0; vg < 8; ++vg) {
      int rl = tr * 16 + vg + 8 * hiL;
      sS[rl * T + j] = acc[vg] * scale + biasTab[(rb0 + rl) * T + j];
    }
  }
  __syncthreads();

  // ---- row softmax ----
  for (int r = tid; r < RB; r += 256) {
    float* row = sS + r * T;
    float m = -1e30f;
    for (int j = 0; j < T; ++j) m = fmaxf(m, row[j]);
    float s = 0.f;
    for (int j = 0; j < T; ++j) { float e = __expf(row[j] - m); row[j] = e; s += e; }
    float inv = 1.f / s;
    for (int j = 0; j < T; ++j) sP[r * TP + j] = (half_t)(row[j] * inv);
  }
  __syncthreads();

  // ---- O = P V ----
  constexpr int CTO = DH / 16;
  for (int tile = wid; tile < RT * CTO; tile += 8) {
    int tr = tile % RT, tc = tile / RT;
    v8f acc = {};
    for (int kk = 0; kk < TP; kk += 32) {
      v16h a = load_a_frag(sP + tr * 16 * TP + kk, TP);
      v16h bb = load_b_frag(sVt + tc * 16 * TP + kk, TP);
      acc = __builtin_amdgcn_wmma_f32_16x16x32_f16(false, a, false, bb, (short)0, acc, false, false);
    }
    int dcol = head * DH + tc * 16 + colL;
#pragma unroll
    for (int vg = 0; vg < 8; ++vg) {
      int t = rb0 + tr * 16 + vg + 8 * hiL;
      int h = bh * WH + t / WWIN, w = bw * WWIN + t % WWIN;
      size_t n = (size_t)(l * HH + h) * WW_ + w;
      attnout[n * DIMC + dcol] = (half_t)acc[vg];
    }
  }
}

// ---------------------------------------------------------------------------
// fusion stage
// ---------------------------------------------------------------------------
__global__ void reduce_g(const half_t* __restrict__ o0, const half_t* __restrict__ o1,
                         const half_t* __restrict__ o2, float* __restrict__ g) {
  int l = blockIdx.x >> 8, c = blockIdx.x & 255;
  float s = 0.f;
  for (int hw = threadIdx.x; hw < HWSZ; hw += 256) {
    size_t off = ((size_t)(l * HWSZ + hw)) * DIMC + c;
    s += (float)o0[off] + (float)o1[off] + (float)o2[off];
  }
  __shared__ float red[256];
  red[threadIdx.x] = s;
  __syncthreads();
  for (int st = 128; st > 0; st >>= 1) {
    if (threadIdx.x < st) red[threadIdx.x] += red[threadIdx.x + st];
    __syncthreads();
  }
  if (threadIdx.x == 0) g[blockIdx.x] = red[0] * (1.f / (float)HWSZ);
}

__global__ void fusion_kernel(const float* __restrict__ g, const float* __restrict__ fc1,
                              const float* __restrict__ lnw, const float* __restrict__ lnb,
                              const float* __restrict__ fc2, float* __restrict__ aOut) {
  int l = blockIdx.x, j = threadIdx.x;
  __shared__ float sg[256], sh[256], red[256];
  sg[j] = g[l * 256 + j];
  __syncthreads();
  float h1 = 0.f;
  for (int k = 0; k < 256; ++k) h1 += sg[k] * fc1[k * 256 + j];
  red[j] = h1; __syncthreads();
  for (int st = 128; st > 0; st >>= 1) { if (j < st) red[j] += red[j + st]; __syncthreads(); }
  float mu = red[0] * (1.f / 256.f);
  __syncthreads();
  float d = h1 - mu;
  red[j] = d * d; __syncthreads();
  for (int st = 128; st > 0; st >>= 1) { if (j < st) red[j] += red[j + st]; __syncthreads(); }
  float var = red[0] * (1.f / 256.f);
  __syncthreads();
  float hn = d * rsqrtf(var + 1e-5f) * lnw[j] + lnb[j];
  sh[j] = fmaxf(hn, 0.f);
  __syncthreads();
  float s0 = 0.f, s1 = 0.f, s2 = 0.f;
  for (int k = 0; k < 256; ++k) {
    float hv = sh[k];
    const float* r = fc2 + k * 768;
    s0 += hv * r[j]; s1 += hv * r[256 + j]; s2 += hv * r[512 + j];
  }
  float m = fmaxf(s0, fmaxf(s1, s2));
  float e0 = __expf(s0 - m), e1 = __expf(s1 - m), e2 = __expf(s2 - m);
  float inv = 1.f / (e0 + e1 + e2);
  aOut[l * 768 + j] = e0 * inv;
  aOut[l * 768 + 256 + j] = e1 * inv;
  aOut[l * 768 + 512 + j] = e2 * inv;
}

__global__ void combine(const half_t* __restrict__ o0, const half_t* __restrict__ o1,
                        const half_t* __restrict__ o2, const float* __restrict__ aBuf,
                        float* __restrict__ out) {
  size_t idx = (size_t)blockIdx.x * 256 + threadIdx.x;   // over L*C*HW
  int hw = (int)(idx % HWSZ);
  size_t r = idx / HWSZ;
  int c = (int)(r % DIMC);
  int l = (int)(r / DIMC);
  size_t off = ((size_t)(l * HWSZ + hw)) * DIMC + c;
  const float* a = aBuf + l * 768;
  out[idx] = (float)o0[off] * a[c] + (float)o1[off] * a[256 + c] + (float)o2[off] * a[512 + c];
}

// ---------------------------------------------------------------------------
static size_t attn_lds(int T, int DH, int RB) {
  int DHP = DH < 32 ? 32 : DH, TP = T < 32 ? 32 : T;
  return (size_t)RB * T * 4 +
         2ull * ((size_t)RB * DHP + 2ull * T * DHP + (size_t)DH * TP + (size_t)RB * TP);
}

extern "C" void kernel_launch(void* const* d_in, const int* in_sizes, int n_in,
                              void* d_out, int out_size, void* d_ws, size_t ws_size,
                              hipStream_t stream) {
  const float* x     = (const float*)d_in[0];
  const float* qkvw[3] = { (const float*)d_in[1], (const float*)d_in[5], (const float*)d_in[9] };
  const float* outw[3] = { (const float*)d_in[2], (const float*)d_in[6], (const float*)d_in[10] };
  const float* outb[3] = { (const float*)d_in[3], (const float*)d_in[7], (const float*)d_in[11] };
  const float* pos[3]  = { (const float*)d_in[4], (const float*)d_in[8], (const float*)d_in[12] };
  const float* fc1 = (const float*)d_in[13];
  const float* lnw = (const float*)d_in[14];
  const float* lnb = (const float*)d_in[15];
  const float* fc2 = (const float*)d_in[16];
  float* out = (float*)d_out;

  char* ws = (char*)d_ws;
  size_t off = 0;
  auto alloc = [&](size_t bytes) { size_t o = off; off = (off + bytes + 255) & ~(size_t)255; return o; };
  half_t* xh      = (half_t*)(ws + alloc((size_t)NTOK * DIMC * 2));
  half_t* qkvb    = (half_t*)(ws + alloc((size_t)NTOK * 768 * 2));
  half_t* attnout = (half_t*)(ws + alloc((size_t)NTOK * DIMC * 2));
  half_t* obuf[3];
  for (int i = 0; i < 3; ++i) obuf[i] = (half_t*)(ws + alloc((size_t)NTOK * DIMC * 2));
  half_t* wqT[3]; half_t* woT[3];   // transposed (N,K) f16 weights
  for (int i = 0; i < 3; ++i) wqT[i] = (half_t*)(ws + alloc((size_t)768 * 256 * 2));
  for (int i = 0; i < 3; ++i) woT[i] = (half_t*)(ws + alloc((size_t)256 * 256 * 2));
  float* btab0 = (float*)(ws + alloc(16 * 16 * 4));
  float* btab1 = (float*)(ws + alloc(64 * 64 * 4));
  float* btab2 = (float*)(ws + alloc(256 * 256 * 4));
  float* gbuf  = (float*)(ws + alloc(512 * 4));
  float* aBuf  = (float*)(ws + alloc(1536 * 4));
  (void)ws_size; (void)in_sizes; (void)n_in; (void)out_size;

  // 1) pack + weight conversion (transposed) + bias tables
  pack_x<<<4096, 256, 0, stream>>>(x, xh);
  for (int i = 0; i < 3; ++i) {
    cvt_f16_t<<<(256 * 768 + 255) / 256, 256, 0, stream>>>(qkvw[i], wqT[i], 256, 768);
    cvt_f16_t<<<(256 * 256 + 255) / 256, 256, 0, stream>>>(outw[i], woT[i], 256, 256);
  }
  build_bias<4, 4><<<(16 * 16 + 255) / 256, 256, 0, stream>>>(pos[0], btab0);
  build_bias<8, 8><<<(64 * 64 + 255) / 256, 256, 0, stream>>>(pos[1], btab1);
  build_bias<16, 16><<<(256 * 256 + 255) / 256, 256, 0, stream>>>(pos[2], btab2);

  // 2) per-branch: qkv GEMM -> window attention -> output projection
  // branch 0: wh=ww=4, heads=16, dh=16
  gemm_f16<<<dim3(768 / 128, NTOK / 128), 256, 0, stream>>>(xh, 256, wqT[0], 256, qkvb, 768, nullptr, 256);
  win_attn<4, 4, 16, 16, 16><<<2 * 2048 * 16, 256, attn_lds(16, 16, 16), stream>>>(
      qkvb, btab0, attnout, 0.25f);
  gemm_f16<<<dim3(256 / 128, NTOK / 128), 256, 0, stream>>>(attnout, 256, woT[0], 256, obuf[0], 256, outb[0], 256);

  // branch 1: wh=ww=8, heads=8, dh=32
  gemm_f16<<<dim3(768 / 128, NTOK / 128), 256, 0, stream>>>(xh, 256, wqT[1], 256, qkvb, 768, nullptr, 256);
  win_attn<8, 8, 8, 32, 64><<<2 * 512 * 8, 256, attn_lds(64, 32, 64), stream>>>(
      qkvb, btab1, attnout, 0.1767766953f);
  gemm_f16<<<dim3(256 / 128, NTOK / 128), 256, 0, stream>>>(attnout, 256, woT[1], 256, obuf[1], 256, outb[1], 256);

  // branch 2: wh=ww=16, heads=4, dh=64; row-blocks of 32
  gemm_f16<<<dim3(768 / 128, NTOK / 128), 256, 0, stream>>>(xh, 256, wqT[2], 256, qkvb, 768, nullptr, 256);
  win_attn<16, 16, 4, 64, 32><<<2 * 128 * 4 * 8, 256, attn_lds(256, 64, 32), stream>>>(
      qkvb, btab2, attnout, 0.125f);
  gemm_f16<<<dim3(256 / 128, NTOK / 128), 256, 0, stream>>>(attnout, 256, woT[2], 256, obuf[2], 256, outb[2], 256);

  // 3) fusion
  reduce_g<<<512, 256, 0, stream>>>(obuf[0], obuf[1], obuf[2], gbuf);
  fusion_kernel<<<2, 256, 0, stream>>>(gbuf, fc1, lnw, lnb, fc2, aBuf);
  combine<<<(int)(((size_t)LL * DIMC * HWSZ) / 256), 256, 0, stream>>>(obuf[0], obuf[1], obuf[2], aBuf, out);
}
